// RNN_46282567582013
// MI455X (gfx1250) — compile-verified, matrix operand order checked by reading, not run on hardware
//
#include <hip/hip_runtime.h>
#include <hip/hip_bf16.h>
#include <math.h>

// ---------------------------------------------------------------------------
// LSTM on MI455X (gfx1250).
//  * Input projection: big bf16-WMMA GEMM (M=16384, N=4096, K=1024).
//  * Recurrent scan: ONE persistent kernel, 128 blocks x 512 threads.
//    - W_hh (8 MB bf16) register-cached across all 512 steps (64 VGPRs/wave).
//    - h_prev tile staged per step with GLOBAL_LOAD_ASYNC_TO_LDS_B128
//      (one 32 KB contiguous async copy, s_wait_asynccnt), WMMA operands
//      then come from LDS instead of 4x-duplicated L2 loads.
//    - Register-resident cell state; atomic grid barrier between steps.
// ---------------------------------------------------------------------------

typedef __bf16 bf16_t;
typedef __attribute__((ext_vector_type(16))) __bf16        v16bf;
typedef __attribute__((ext_vector_type(8)))  float         v8f;
typedef __attribute__((ext_vector_type(4)))  unsigned int  u32x4;

constexpr int B  = 32;
constexpr int T  = 512;
constexpr int I  = 1024;
constexpr int H  = 1024;
constexpr int G4 = 4096;   // 4*H

union ABFrag { v16bf v; u32x4 q[2]; };

// Load one 16x32 bf16 WMMA operand fragment from a row-major matrix.
// 16-bit operand layout (ISA 7.12.2): lanes 0-15 -> rows 0-15, kbase=0;
// lanes 16-31 -> rows 0-15, kbase=8.  Elements 0-7 hold K=kbase..kbase+7,
// elements 8-15 hold K=kbase+16..kbase+23  => two contiguous 16B loads.
static __device__ __forceinline__
void load_frag(ABFrag& f, const bf16_t* __restrict__ base, int rowbase,
               int ldk, int kofs) {
  const int ln = threadIdx.x & 31;
  const bf16_t* p = base + (size_t)(rowbase + (ln & 15)) * ldk
                         + kofs + ((ln >> 4) << 3);
  f.q[0] = *reinterpret_cast<const u32x4*>(p);
  f.q[1] = *reinterpret_cast<const u32x4*>(p + 16);
}

static __device__ __forceinline__
v8f wmma_bf16(const ABFrag& a, const ABFrag& b, v8f c) {
  return __builtin_amdgcn_wmma_f32_16x16x32_bf16(
      false, a.v, false, b.v, (short)0, c, false, false);
}

// ------------------------- small utility kernels ---------------------------
__global__ void __launch_bounds__(256)
convert_f32_bf16(const float* __restrict__ src, bf16_t* __restrict__ dst, int n) {
  int i = blockIdx.x * 256 + threadIdx.x;
  if (i < n) dst[i] = (bf16_t)src[i];
}

__global__ void __launch_bounds__(256)
zero_u32(unsigned* __restrict__ dst, int n) {
  int i = blockIdx.x * 256 + threadIdx.x;
  if (i < n) dst[i] = 0u;
}

// --------------------- xW = x . W_ih^T + b  (big GEMM) ---------------------
__global__ void __launch_bounds__(256)
gemm_xw_kernel(const bf16_t* __restrict__ xbf, const bf16_t* __restrict__ wih,
               const float* __restrict__ bias, float* __restrict__ xW) {
  const int w     = threadIdx.x >> 5;
  const int ln    = threadIdx.x & 31;
  const int mbase = (blockIdx.y * 8 + w) * 16;
  const int nbase = blockIdx.x * 64;

  v8f acc[4] = {};
  ABFrag a0, bcur[4];
  load_frag(a0, xbf, mbase, I, 0);
#pragma unroll
  for (int j = 0; j < 4; ++j) load_frag(bcur[j], wih, nbase + j * 16, I, 0);

  for (int kc = 0; kc < I; kc += 32) {
    const int knext = (kc + 32 < I) ? kc + 32 : 0;  // branch-free epilogue
    ABFrag a1, bnext[4];
    load_frag(a1, xbf, mbase, I, knext);
#pragma unroll
    for (int j = 0; j < 4; ++j)
      load_frag(bnext[j], wih, nbase + j * 16, I, knext);
#pragma unroll
    for (int j = 0; j < 4; ++j)
      acc[j] = wmma_bf16(a0, bcur[j], acc[j]);
    a0 = a1;
#pragma unroll
    for (int j = 0; j < 4; ++j) bcur[j] = bnext[j];
  }

  // C/D layout: VGPR r, lanes 0-15 -> M=r, lanes 16-31 -> M=8+r; N = lane&15.
  const int ncol = ln & 15;
  const int mlo  = (ln >> 4) << 3;
#pragma unroll
  for (int j = 0; j < 4; ++j) {
    const int n  = nbase + j * 16 + ncol;
    const float bj = bias[n];
#pragma unroll
    for (int r = 0; r < 8; ++r)
      xW[(size_t)(mbase + mlo + r) * G4 + n] = acc[j][r] + bj;
  }
}

// ----------------------- persistent recurrent scan -------------------------
// 128 blocks: blockIdx.x = (hcol<<1)|mtile; block owns one 16x16 (batch x
// hidden) tile for ALL 512 steps.  512 threads = 16 waves; wave w handles
// gate w&3 over K-quarter w>>2.  W_hh fragments live in registers; h_prev
// is staged into LDS each step with async global->LDS copies.
__global__ void __launch_bounds__(512)
lstm_scan_kernel(const float* __restrict__ xW, const bf16_t* __restrict__ whh,
                 bf16_t* __restrict__ hb0, bf16_t* __restrict__ hb1,
                 float* __restrict__ out, unsigned* __restrict__ sync) {
  __shared__ float  zls[16][256];        // 16 KB partial-z reduction
  __shared__ bf16_t hstage[16 * H];      // 32 KB staged h_prev tile

  const int w     = threadIdx.x >> 5;
  const int ln    = threadIdx.x & 31;
  const int gate  = w & 3;
  const int kq    = w >> 2;                 // K-quarter 0..3
  const int mtile = blockIdx.x & 1;
  const int hcol  = blockIdx.x >> 1;
  const int mbase = mtile * 16;
  const int nrow  = gate * H + hcol * 16;   // W_hh row block (== z columns)
  const int k0    = kq * (H / 4);           // this wave's K range: [k0, k0+256)

  // ---- persistent register cache of this wave's W_hh slice (8 KB) ----
  ABFrag bc[8];
#pragma unroll
  for (int i = 0; i < 8; ++i) load_frag(bc[i], whh, nrow, H, k0 + i * 32);

  // Async-copy operands (invariant): LDS byte address (generic-pointer low
  // 32 bits == LDS offset per the flat-aperture mapping) and global offset.
  const unsigned lofs = (unsigned)(size_t)(&hstage[0]) + threadIdx.x * 16u;
  const unsigned gofs = threadIdx.x * 16u;
  const bf16_t* hs = hstage;

  // Element owned by lanes 0..255 in the update phase.
  const int m   = threadIdx.x >> 4;
  const int n   = threadIdx.x & 15;
  const int row = mbase + m;          // batch index
  const int col = hcol * 16 + n;      // hidden index
  const size_t co = (size_t)row * H + col;
  float creg = 0.0f;                  // cell state lives in a register

  for (int t = 0; t < T; ++t) {
    const bf16_t* __restrict__ hsrc =
        ((t & 1) ? hb1 : hb0) + (size_t)mbase * H;   // 32 KB contiguous
    bf16_t* __restrict__ hnext = (t & 1) ? hb0 : hb1;

    // ---- cooperative async fill: 32 KB global -> LDS (ASYNCcnt path).
    // GVS addressing: the immediate offset adds to both LDS lofs and global saddr+gofs, so the
    // same offset immediate walks both sides in 8 KB strides.
    asm volatile(
        "global_load_async_to_lds_b128 %0, %1, %2\n\t"
        "global_load_async_to_lds_b128 %0, %1, %2 offset:8192\n\t"
        "global_load_async_to_lds_b128 %0, %1, %2 offset:16384\n\t"
        "global_load_async_to_lds_b128 %0, %1, %2 offset:24576"
        :: "v"(lofs), "v"(gofs), "s"(hsrc) : "memory");
    asm volatile("s_wait_asynccnt 0x0" ::: "memory");
    __syncthreads();

    // ---- z partial tile: 8 WMMAs fed from LDS (rows 0..15 == tile rows) ----
    v8f acc = {};
    ABFrag a0;
    load_frag(a0, hs, 0, H, k0);
#pragma unroll
    for (int i = 0; i < 8; ++i) {
      ABFrag a1;
      load_frag(a1, hs, 0, H, k0 + ((i + 1) & 7) * 32);
      acc = wmma_bf16(a0, bc[i], acc);
      a0 = a1;
    }

    const int ncol = ln & 15;
    const int mlo  = (ln >> 4) << 3;
#pragma unroll
    for (int r = 0; r < 8; ++r)
      zls[w][(mlo + r) * 16 + ncol] = acc[r];
    __syncthreads();   // also fences hstage reads before next-step overwrite

    // ---- cell update: lanes 0..255, one per tile element ----
    if (threadIdx.x < 256) {
      const int e = threadIdx.x;      // elem idx == m*16+n
      const size_t xo = ((size_t)row * T + t) * G4 + col;
      const float zi = zls[0][e] + zls[4][e] + zls[8][e]  + zls[12][e] + xW[xo];
      const float zf = zls[1][e] + zls[5][e] + zls[9][e]  + zls[13][e] + xW[xo + H];
      const float zg = zls[2][e] + zls[6][e] + zls[10][e] + zls[14][e] + xW[xo + 2 * H];
      const float zo = zls[3][e] + zls[7][e] + zls[11][e] + zls[15][e] + xW[xo + 3 * H];

      const float ig = 1.0f / (1.0f + __expf(-zi));
      const float fg = 1.0f / (1.0f + __expf(-zf));
      const float gg = tanhf(zg);
      const float og = 1.0f / (1.0f + __expf(-zo));

      creg = fg * creg + ig * gg;
      const float hn = og * tanhf(creg);

      out[((size_t)row * T + t) * H + col] = hn;
      hnext[co] = (bf16_t)hn;
    }

    // ---- device-scope grid barrier between steps ----
    __threadfence();
    __syncthreads();
    if (threadIdx.x == 0) {
      const unsigned target = (unsigned)(t + 1) * gridDim.x;
      __hip_atomic_fetch_add(sync, 1u, __ATOMIC_RELEASE,
                             __HIP_MEMORY_SCOPE_AGENT);
      while (__hip_atomic_load(sync, __ATOMIC_ACQUIRE,
                               __HIP_MEMORY_SCOPE_AGENT) < target)
        __builtin_amdgcn_s_sleep(1);
    }
    __syncthreads();
  }
}

// --------------------------------- launch ----------------------------------
extern "C" void kernel_launch(void* const* d_in, const int* in_sizes, int n_in,
                              void* d_out, int out_size, void* d_ws, size_t ws_size,
                              hipStream_t stream) {
  (void)in_sizes; (void)n_in; (void)out_size; (void)ws_size;

  const float* x    = (const float*)d_in[0];   // [B,T,I]
  const float* Wih  = (const float*)d_in[1];   // [4H,I]
  const float* Whh  = (const float*)d_in[2];   // [4H,H]
  const float* bias = (const float*)d_in[3];   // [4H]
  float* out = (float*)d_out;                  // [B,T,H]

  // Workspace carve-up.
  char* ws = (char*)d_ws;
  float*    xW   = (float*)ws;    ws += (size_t)B * T * G4 * sizeof(float); // 268 MB
  bf16_t*   xbf  = (bf16_t*)ws;   ws += (size_t)B * T * I * sizeof(bf16_t); //  33 MB
  bf16_t*   wihb = (bf16_t*)ws;   ws += (size_t)G4 * I * sizeof(bf16_t);    //   8 MB
  bf16_t*   whhb = (bf16_t*)ws;   ws += (size_t)G4 * H * sizeof(bf16_t);    //   8 MB
  bf16_t*   hb0  = (bf16_t*)ws;   ws += (size_t)B * H * sizeof(bf16_t);     //  64 KB
  bf16_t*   hb1  = (bf16_t*)ws;   ws += (size_t)B * H * sizeof(bf16_t);     //  64 KB
  unsigned* sync = (unsigned*)ws;                                           // barrier ctr

  // 1) bf16 conversions; zero h0 and the barrier counter (fresh every call).
  const int nx = B * T * I;
  convert_f32_bf16<<<(nx + 255) / 256, 256, 0, stream>>>(x, xbf, nx);
  convert_f32_bf16<<<(G4 * I + 255) / 256, 256, 0, stream>>>(Wih, wihb, G4 * I);
  convert_f32_bf16<<<(G4 * H + 255) / 256, 256, 0, stream>>>(Whh, whhb, G4 * H);
  zero_u32<<<(B * H / 2 + 255) / 256, 256, 0, stream>>>((unsigned*)hb0, B * H / 2);
  zero_u32<<<1, 256, 0, stream>>>(sync, 1);

  // 2) Input projection for all timesteps (parallel WMMA GEMM).
  gemm_xw_kernel<<<dim3(64, 128), 256, 0, stream>>>(xbf, wihb, bias, xW);

  // 3) Whole recurrence in one persistent kernel.
  lstm_scan_kernel<<<128, 512, 0, stream>>>(xW, whhb, hb0, hb1, out, sync);
}
